// MSLE_22531398434985
// MI455X (gfx1250) — matrix operand order repaired and checked by reading, not compiled, not run on hardware
//
#include <hip/hip_runtime.h>
#include <hip/hip_bf16.h>
#include <stddef.h>

// Problem constants (match reference)
#define XPER   1024      // gathered elements per sample (ARRAY*EMB)
#define SPAT   512       // 8*8*8
#define OC_    16
#define FCIN   24576     // 3*16*512
#define EMB_   512
#define RED_   32

typedef __attribute__((ext_vector_type(16))) __bf16 v16bf;
typedef __attribute__((ext_vector_type(8)))  float  v8f;

__device__ __forceinline__ unsigned short f32_to_bf16(float f) {
  unsigned int u = __float_as_uint(f);
  unsigned int r = u + 0x7FFFu + ((u >> 16) & 1u);   // round-to-nearest-even
  return (unsigned short)(r >> 16);
}

// CDNA5 async global->LDS copy (ASYNCcnt-tracked). Per-lane 16B copy:
// LDS[lds_byte_addr] = MEM[gptr..gptr+15]. GV addressing mode.
__device__ __forceinline__ void async_copy_b128(unsigned lds_byte_addr, const void* gptr) {
  asm volatile("global_load_async_to_lds_b128 %0, %1, off"
               :: "v"(lds_byte_addr), "v"(gptr)
               : "memory");
}
__device__ __forceinline__ void wait_asynccnt0() {
  asm volatile("s_wait_asynccnt 0x0" ::: "memory");
}

// x layout per sample: xs[c*512 + d*64 + h*8 + w], zero beyond index 7 (high-side SAME pad)
__device__ __forceinline__ float xat(const float* xs, int c, int d, int h, int w) {
  return (d < 8 && h < 8 && w < 8) ? xs[c * 512 + d * 64 + h * 8 + w] : 0.0f;
}

// filt layout: f1 at [0..63] (oc*4+ci*2+kw), f3 at [64..191] (oc*8+ci*4+kh*2+kw),
// f5 at [192..447] (oc*16+ci*8+kd*4+kh*2+kw)
__device__ __forceinline__ float conv_eval(const float* xs, const float* filt,
                                           int br, int oc, int d, int h, int w) {
  float y = 0.0f;
  if (br == 0) {
    const float* f = filt + oc * 4;
#pragma unroll
    for (int ci = 0; ci < 2; ++ci)
#pragma unroll
      for (int kw = 0; kw < 2; ++kw)
        y += xat(xs, ci, d, h, w + kw) * f[ci * 2 + kw];
  } else if (br == 1) {
    const float* f = filt + 64 + oc * 8;
#pragma unroll
    for (int ci = 0; ci < 2; ++ci)
#pragma unroll
      for (int kh = 0; kh < 2; ++kh)
#pragma unroll
        for (int kw = 0; kw < 2; ++kw)
          y += xat(xs, ci, d, h + kh, w + kw) * f[ci * 4 + kh * 2 + kw];
  } else {
    const float* f = filt + 192 + oc * 16;
#pragma unroll
    for (int ci = 0; ci < 2; ++ci)
#pragma unroll
      for (int kd = 0; kd < 2; ++kd)
#pragma unroll
        for (int kh = 0; kh < 2; ++kh)
#pragma unroll
          for (int kw = 0; kw < 2; ++kw)
            y += xat(xs, ci, d + kd, h + kh, w + kw) * f[ci * 8 + kd * 4 + kh * 2 + kw];
  }
  return y;
}

// ---------------------------------------------------------------------------
// K0: chequer gather + BN0 batch statistics (per-channel sum/sumsq, 2 channels)
// ---------------------------------------------------------------------------
__global__ __launch_bounds__(256)
void msle_gather_stats(const float* __restrict__ ci, const int* __restrict__ perm,
                       float* __restrict__ xg, float* __restrict__ stats) {
  const int b = blockIdx.x, tid = threadIdx.x;
  float s0 = 0.f, q0 = 0.f, s1 = 0.f, q1 = 0.f;
#pragma unroll
  for (int k = 0; k < 4; ++k) {
    int i = tid + k * 256;
    float v = ci[(size_t)b * XPER + perm[i]];
    xg[(size_t)b * XPER + i] = v;
    if (i < 512) { s0 += v; q0 += v * v; } else { s1 += v; q1 += v * v; }
  }
  __shared__ float ss[4];
  if (tid < 4) ss[tid] = 0.f;
  __syncthreads();
  atomicAdd(&ss[0], s0); atomicAdd(&ss[1], q0);
  atomicAdd(&ss[2], s1); atomicAdd(&ss[3], q1);
  __syncthreads();
  if (tid < 4) atomicAdd(&stats[tid], ss[tid]);
}

// ---------------------------------------------------------------------------
// K2: per-sample convs -> BN1/2/3 channel stats + per-sample spatial sums (SE)
// ---------------------------------------------------------------------------
__global__ __launch_bounds__(256)
void msle_conv_stats(const float* __restrict__ xg, const int* __restrict__ rel,
                     const float* __restrict__ f1t, const float* __restrict__ f3t,
                     const float* __restrict__ f5t,
                     const float* __restrict__ bn0g, const float* __restrict__ bn0b,
                     const float* __restrict__ stats,
                     float* __restrict__ chsum, float* __restrict__ chsq,
                     float* __restrict__ bsum, int B) {
  __shared__ float xs[1024];
  __shared__ float filt[448];
  __shared__ float ssum[48], ssq[48];
  const int b = blockIdx.x, tid = threadIdx.x;
  const float cnt = (float)B * 512.0f;
  const float m0 = stats[0] / cnt, v0 = stats[1] / cnt - m0 * m0;
  const float m1 = stats[2] / cnt, v1 = stats[3] / cnt - m1 * m1;
  const float sc0 = bn0g[0] * rsqrtf(v0 + 1e-5f), sh0 = bn0b[0] - m0 * sc0;
  const float sc1 = bn0g[1] * rsqrtf(v1 + 1e-5f), sh1 = bn0b[1] - m1 * sc1;
#pragma unroll
  for (int k = 0; k < 4; ++k) {
    int i = tid + k * 256;
    float v = xg[(size_t)b * XPER + i];
    xs[i] = (i < 512) ? (v * sc0 + sh0) : (v * sc1 + sh1);
  }
  const int r = rel[b];
  if (tid < 64)  filt[tid]       = f1t[(size_t)r * 64  + tid];
  if (tid < 128) filt[64 + tid]  = f3t[(size_t)r * 128 + tid];
  filt[192 + tid] = f5t[(size_t)r * 256 + tid];
  if (tid < 48) { ssum[tid] = 0.f; ssq[tid] = 0.f; }
  __syncthreads();

  for (int p = tid; p < 3 * OC_ * SPAT; p += 256) {
    int br = p >> 13, rem = p & 8191;
    int oc = rem >> 9, s = rem & 511;
    int d = s >> 6, h = (s >> 3) & 7, w = s & 7;
    float y = conv_eval(xs, filt, br, oc, d, h, w);
    atomicAdd(&ssum[br * 16 + oc], y);
    atomicAdd(&ssq[br * 16 + oc], y * y);
  }
  __syncthreads();
  if (tid < 48) {
    bsum[(size_t)b * 48 + tid] = ssum[tid];
    atomicAdd(&chsum[tid], ssum[tid]);
    atomicAdd(&chsq[tid], ssq[tid]);
  }
}

// ---------------------------------------------------------------------------
// K3: SE attention gates (one thread per sample)
// ---------------------------------------------------------------------------
__global__ __launch_bounds__(256)
void msle_gates(const float* __restrict__ bsum,
                const float* __restrict__ chsum, const float* __restrict__ chsq,
                const float* __restrict__ bn1g, const float* __restrict__ bn1b,
                const float* __restrict__ bn2g, const float* __restrict__ bn2b,
                const float* __restrict__ bn3g, const float* __restrict__ bn3b,
                const float* __restrict__ Wsq, const float* __restrict__ bsq,
                const float* __restrict__ W1, const float* __restrict__ b1,
                const float* __restrict__ W2, const float* __restrict__ b2,
                const float* __restrict__ W3, const float* __restrict__ b3,
                float* __restrict__ gates, int B) {
  const int b = blockIdx.x * 256 + threadIdx.x;
  if (b >= B) return;
  const float cnt = (float)B * 512.0f;
  const float* gs[3] = { bn1g, bn2g, bn3g };
  const float* bs[3] = { bn1b, bn2b, bn3b };

  float pooled[16];
#pragma unroll
  for (int oc = 0; oc < 16; ++oc) pooled[oc] = 0.f;
#pragma unroll
  for (int br = 0; br < 3; ++br)
#pragma unroll
    for (int oc = 0; oc < 16; ++oc) {
      int c = br * 16 + oc;
      float m = chsum[c] / cnt, v = chsq[c] / cnt - m * m;
      float sc = gs[br][oc] * rsqrtf(v + 1e-5f);
      float sh = bs[br][oc] - m * sc;
      pooled[oc] += bsum[(size_t)b * 48 + c] * (1.0f / 512.0f) * sc + sh;
    }
  float zq[32];
#pragma unroll
  for (int j = 0; j < 32; ++j) {
    float a = bsq[j];
#pragma unroll
    for (int oc = 0; oc < 16; ++oc) a += pooled[oc] * Wsq[oc * 32 + j];
    zq[j] = fmaxf(a, 0.f);
  }
  const float* Ws[3] = { W1, W2, W3 };
  const float* Wb[3] = { b1, b2, b3 };
#pragma unroll
  for (int br = 0; br < 3; ++br)
#pragma unroll
    for (int oc = 0; oc < 16; ++oc) {
      float a = Wb[br][oc];
#pragma unroll
      for (int j = 0; j < 32; ++j) a += zq[j] * Ws[br][j * 16 + oc];
      gates[(size_t)b * 48 + br * 16 + oc] = 1.0f / (1.0f + expf(-a));
    }
}

// ---------------------------------------------------------------------------
// K4: recompute convs, apply BN + gate + ReLU, emit bf16 activations (B x 24576)
// ---------------------------------------------------------------------------
__global__ __launch_bounds__(256)
void msle_act(const float* __restrict__ xg, const int* __restrict__ rel,
              const float* __restrict__ f1t, const float* __restrict__ f3t,
              const float* __restrict__ f5t,
              const float* __restrict__ bn0g, const float* __restrict__ bn0b,
              const float* __restrict__ stats,
              const float* __restrict__ chsum, const float* __restrict__ chsq,
              const float* __restrict__ bn1g, const float* __restrict__ bn1b,
              const float* __restrict__ bn2g, const float* __restrict__ bn2b,
              const float* __restrict__ bn3g, const float* __restrict__ bn3b,
              const float* __restrict__ gates,
              unsigned short* __restrict__ act, int B) {
  __shared__ float xs[1024];
  __shared__ float filt[448];
  __shared__ float sc[48], sh[48], gt[48];
  const int b = blockIdx.x, tid = threadIdx.x;
  const float cnt = (float)B * 512.0f;
  const float m0 = stats[0] / cnt, v0 = stats[1] / cnt - m0 * m0;
  const float m1 = stats[2] / cnt, v1 = stats[3] / cnt - m1 * m1;
  const float s0 = bn0g[0] * rsqrtf(v0 + 1e-5f), o0 = bn0b[0] - m0 * s0;
  const float s1 = bn0g[1] * rsqrtf(v1 + 1e-5f), o1 = bn0b[1] - m1 * s1;
#pragma unroll
  for (int k = 0; k < 4; ++k) {
    int i = tid + k * 256;
    float v = xg[(size_t)b * XPER + i];
    xs[i] = (i < 512) ? (v * s0 + o0) : (v * s1 + o1);
  }
  const int r = rel[b];
  if (tid < 64)  filt[tid]       = f1t[(size_t)r * 64  + tid];
  if (tid < 128) filt[64 + tid]  = f3t[(size_t)r * 128 + tid];
  filt[192 + tid] = f5t[(size_t)r * 256 + tid];
  if (tid < 48) {
    int br = tid >> 4, oc = tid & 15;
    float m = chsum[tid] / cnt, v = chsq[tid] / cnt - m * m;
    const float* gptr = (br == 0) ? bn1g : ((br == 1) ? bn2g : bn3g);
    const float* bptr = (br == 0) ? bn1b : ((br == 1) ? bn2b : bn3b);
    float s = gptr[oc] * rsqrtf(v + 1e-5f);
    sc[tid] = s; sh[tid] = bptr[oc] - m * s;
    gt[tid] = gates[(size_t)b * 48 + tid];
  }
  __syncthreads();

  for (int p = tid; p < 3 * OC_ * SPAT; p += 256) {
    int br = p >> 13, rem = p & 8191;
    int oc = rem >> 9, s = rem & 511;
    int d = s >> 6, h = (s >> 3) & 7, w = s & 7;
    float y = conv_eval(xs, filt, br, oc, d, h, w);
    int c = br * 16 + oc;
    float val = fmaxf((y * sc[c] + sh[c]) * gt[c], 0.0f);
    act[(size_t)b * FCIN + p] = f32_to_bf16(val);
  }
}

// ---------------------------------------------------------------------------
// K5: fc_W fp32 -> bf16
// ---------------------------------------------------------------------------
__global__ __launch_bounds__(256)
void msle_cvt_w(const float* __restrict__ w, unsigned short* __restrict__ wb, int n4) {
  int i = blockIdx.x * 256 + threadIdx.x;
  if (i >= n4) return;
  float4 v = ((const float4*)w)[i];
  ushort4 o;
  o.x = f32_to_bf16(v.x); o.y = f32_to_bf16(v.y);
  o.z = f32_to_bf16(v.z); o.w = f32_to_bf16(v.w);
  ((ushort4*)wb)[i] = o;
}

// ---------------------------------------------------------------------------
// K6: out(B,512) = act(B,24576) @ W(24576,512) + bias   [bf16 WMMA, f32 acc]
// Block tile 128x128, 8 waves, each wave = 2x4 tiles of 16x16, K-chunk 32.
// Double-buffered LDS; A tile staged with CDNA5 async global->LDS copies
// (ASYNCcnt), B tile staged via regs with an in-LDS transpose; one barrier
// per K-chunk so next-chunk fills overlap the current chunk's WMMAs.
// ---------------------------------------------------------------------------
__global__ __launch_bounds__(256)
void msle_gemm_bf16(const unsigned short* __restrict__ act,
                    const unsigned short* __restrict__ wb,
                    const float* __restrict__ bias,
                    float* __restrict__ out, int K) {
  __shared__ __align__(16) unsigned short Atile[2][128 * 40]; // [m][k], stride 40
  __shared__ __align__(16) unsigned short Btile[2][128 * 40]; // [n][k], stride 40

  const int tid  = threadIdx.x;
  const int lane = tid & 31;
  const int wv   = tid >> 5;
  const int wm   = wv & 3;          // 4 wave-rows of 32
  const int wn   = wv >> 2;         // 2 wave-cols of 64
  const int mbase = blockIdx.x * 128;
  const int nbase = blockIdx.y * 128;

  const int ar  = tid >> 1;         // A row 0..127
  const int ah  = (tid & 1) * 16;   // A col half (16 bf16)
  const int bkk = tid >> 3;         // B k-row 0..31
  const int bnb = (tid & 7) * 16;   // B col group (16 cols)

  v8f acc[2][4];
#pragma unroll
  for (int mt = 0; mt < 2; ++mt)
#pragma unroll
    for (int nt = 0; nt < 4; ++nt)
#pragma unroll
      for (int e = 0; e < 8; ++e) acc[mt][nt][e] = 0.0f;

  const size_t arow = (size_t)(mbase + ar) * (size_t)K + ah;
  const int ml = lane & 15;
  const int hh = lane >> 4;

  // Per-thread LDS byte addresses of A staging slots (low 32 bits of the
  // flat LDS address are the LDS byte offset on CDNA5).
  const unsigned aLds[2] = {
    (unsigned)(uintptr_t)&Atile[0][ar * 40 + ah],
    (unsigned)(uintptr_t)&Atile[1][ar * 40 + ah]
  };

  // Prologue: async A chunk 0 -> buf 0; B chunk 0 -> regs.
  async_copy_b128(aLds[0],      act + arow);
  async_copy_b128(aLds[0] + 16, act + arow + 8);
  const uint4* pb0 = (const uint4*)(wb + (size_t)bkk * EMB_ + nbase + bnb);
  uint4 w0 = pb0[0], w1 = pb0[1];

  const int iters = K / 32;
  for (int it = 0; it < iters; ++it) {
    const int cur = it & 1;
    const int nxt = cur ^ 1;
    const int k0 = it * 32;

    // Stage B chunk into LDS[cur] (transpose to [n][k]).
    union { uint4 u[2]; unsigned short s[16]; } tw;
    tw.u[0] = w0; tw.u[1] = w1;
#pragma unroll
    for (int j = 0; j < 16; ++j) Btile[cur][(bnb + j) * 40 + bkk] = tw.s[j];

    wait_asynccnt0();   // this wave's A-copies into buf[cur] are in LDS
    __syncthreads();    // all waves' fills done; all reads of buf[nxt] done

    if (it + 1 < iters) {
      // Kick off next chunk: async A into buf[nxt], B into regs.
      async_copy_b128(aLds[nxt],      act + arow + k0 + 32);
      async_copy_b128(aLds[nxt] + 16, act + arow + k0 + 40);
      const uint4* pb = (const uint4*)(wb + (size_t)(k0 + 32 + bkk) * EMB_ + nbase + bnb);
      w0 = pb[0]; w1 = pb[1];
      if (it + 2 < iters)
        __builtin_prefetch(wb + (size_t)(k0 + 64 + bkk) * EMB_ + nbase + bnb, 0, 1);
    }

    // A fragment: lanes 0-15 get K pairs {0..7,16..23}, lanes 16-31 {8..15,24..31}
    v16bf afr[2], bfr[4];
#pragma unroll
    for (int mt = 0; mt < 2; ++mt) {
      int m = wm * 32 + mt * 16 + ml;
      union { v16bf v; uint4 u[2]; } fa;
      fa.u[0] = *(const uint4*)&Atile[cur][m * 40 + hh * 8];
      fa.u[1] = *(const uint4*)&Atile[cur][m * 40 + 16 + hh * 8];
      afr[mt] = fa.v;
    }
    // B fragment: lane holds N=lane%16, lanes 0-15 get K {0..7,16..23}, lanes 16-31 {8..15,24..31}
#pragma unroll
    for (int nt = 0; nt < 4; ++nt) {
      int n = wn * 64 + nt * 16 + ml;
      union { v16bf v; uint4 u[2]; } fb;
      fb.u[0] = *(const uint4*)&Btile[cur][n * 40 + hh * 8];
      fb.u[1] = *(const uint4*)&Btile[cur][n * 40 + 16 + hh * 8];
      bfr[nt] = fb.v;
    }
#pragma unroll
    for (int mt = 0; mt < 2; ++mt)
#pragma unroll
      for (int nt = 0; nt < 4; ++nt)
        acc[mt][nt] = __builtin_amdgcn_wmma_f32_16x16x32_bf16(
            false, afr[mt], false, bfr[nt], (short)0, acc[mt][nt], false, false);
  }

  // C/D layout: element e -> M = e + 8*(lane/16), N = lane%16
#pragma unroll
  for (int mt = 0; mt < 2; ++mt) {
    int row0 = mbase + wm * 32 + mt * 16 + hh * 8;
#pragma unroll
    for (int nt = 0; nt < 4; ++nt) {
      int col = nbase + wn * 64 + nt * 16 + ml;
      float bv = bias[col];
#pragma unroll
      for (int e = 0; e < 8; ++e)
        out[(size_t)(row0 + e) * EMB_ + col] = acc[mt][nt][e] + bv;
    }
  }
}

// ---------------------------------------------------------------------------
extern "C" void kernel_launch(void* const* d_in, const int* in_sizes, int n_in,
                              void* d_out, int out_size, void* d_ws, size_t ws_size,
                              hipStream_t stream) {
  const float* concat_input = (const float*)d_in[0];
  const int*   rel  = (const int*)d_in[1];
  const int*   perm = (const int*)d_in[2];
  const float* f1t  = (const float*)d_in[3];
  const float* f3t  = (const float*)d_in[4];
  const float* f5t  = (const float*)d_in[5];
  const float* bn0g = (const float*)d_in[6],  *bn0b = (const float*)d_in[7];
  const float* bn1g = (const float*)d_in[8],  *bn1b = (const float*)d_in[9];
  const float* bn2g = (const float*)d_in[10], *bn2b = (const float*)d_in[11];
  const float* bn3g = (const float*)d_in[12], *bn3b = (const float*)d_in[13];
  const float* Wsq  = (const float*)d_in[14], *bsq  = (const float*)d_in[15];
  const float* W1   = (const float*)d_in[16], *b1   = (const float*)d_in[17];
  const float* W2   = (const float*)d_in[18], *b2   = (const float*)d_in[19];
  const float* W3   = (const float*)d_in[20], *b3   = (const float*)d_in[21];
  const float* fcW  = (const float*)d_in[22], *fcb  = (const float*)d_in[23];
  float* out = (float*)d_out;

  const int B = in_sizes[0] / XPER;   // 4096

  // Workspace carve-up (all regions 16B-aligned):
  //  [0,512):  stats (4) | chsum (48) | chsq (48)   -- zeroed each call
  //  bsum:   B*48 f32, gates: B*48 f32, xg: B*1024 f32,
  //  actb:   B*24576 bf16, wbf: 24576*512 bf16
  char* ws = (char*)d_ws;
  float* stats = (float*)ws;
  float* chsum = stats + 4;
  float* chsq  = chsum + 48;
  float* bsum  = (float*)(ws + 512);
  float* gates = bsum + (size_t)B * 48;
  float* xg    = gates + (size_t)B * 48;
  unsigned short* actb = (unsigned short*)(xg + (size_t)B * XPER);
  unsigned short* wbf  = actb + (size_t)B * FCIN;

  hipMemsetAsync(d_ws, 0, 512, stream);

  msle_gather_stats<<<B, 256, 0, stream>>>(concat_input, perm, xg, stats);
  msle_conv_stats<<<B, 256, 0, stream>>>(xg, rel, f1t, f3t, f5t, bn0g, bn0b,
                                         stats, chsum, chsq, bsum, B);
  msle_gates<<<(B + 255) / 256, 256, 0, stream>>>(bsum, chsum, chsq,
                                                  bn1g, bn1b, bn2g, bn2b, bn3g, bn3b,
                                                  Wsq, bsq, W1, b1, W2, b2, W3, b3,
                                                  gates, B);
  msle_act<<<B, 256, 0, stream>>>(xg, rel, f1t, f3t, f5t, bn0g, bn0b, stats,
                                  chsum, chsq, bn1g, bn1b, bn2g, bn2b, bn3g, bn3b,
                                  gates, actb, B);
  const int n4 = FCIN * EMB_ / 4;
  msle_cvt_w<<<(n4 + 255) / 256, 256, 0, stream>>>(fcW, wbf, n4);

  dim3 gg(B / 128, EMB_ / 128);
  msle_gemm_bf16<<<gg, 256, 0, stream>>>(actb, wbf, fcb, out, FCIN);
}
